// BertTinyFlatten_45243185496275
// MI455X (gfx1250) — matrix operand order repaired and checked by reading, not compiled, not run on hardware
//
#include <hip/hip_runtime.h>
#include <hip/hip_bf16.h>

// ---------------------------------------------------------------------------
// BERT-tiny flattened forward on gfx1250 (MI455X), bf16 WMMA everywhere.
// Double-buffered LDS staging using CDNA5 GLOBAL_LOAD_ASYNC_TO_LDS_B128.
// ---------------------------------------------------------------------------

typedef __attribute__((ext_vector_type(16))) __bf16 v16bf;
typedef __attribute__((ext_vector_type(8)))  __bf16 v8bf;
typedef __attribute__((ext_vector_type(8)))  float  v8f;

#define B_   8
#define S_   1024
#define HID_ 512
#define NH_  8
#define INTER_ 2048

static __device__ __forceinline__ __bf16 tobf(float f) {
  unsigned u = __builtin_bit_cast(unsigned, f);
  u += 0x7FFFu + ((u >> 16) & 1u);              // round-to-nearest-even
  unsigned short h = (unsigned short)(u >> 16);
  return __builtin_bit_cast(__bf16, h);
}

// CDNA5 async copy: 16B per lane, global -> LDS, tracked with ASYNCcnt.
// LDS operand VGPR holds the wave-relative LDS byte address == low 32 bits of
// the generic pointer (flat LDS aperture: addr[31:0] is the LDS offset).
static __device__ __forceinline__ void async_copy_b128(const void* gsrc, void* ldsdst) {
  unsigned lds = (unsigned)(size_t)ldsdst;
  unsigned long long ga = (unsigned long long)(size_t)gsrc;
  asm volatile("global_load_async_to_lds_b128 %0, %1, off"
               :: "v"(lds), "v"(ga) : "memory");
}
static __device__ __forceinline__ void wait_async0() {
  asm volatile("s_wait_asynccnt 0x0" ::: "memory");
}

// ------------------------- fp32 -> bf16 converter --------------------------
__global__ __launch_bounds__(256) void f32_to_bf16_kernel(const float* __restrict__ in,
                                                          __bf16* __restrict__ out,
                                                          long long n) {
  long long i = (long long)blockIdx.x * blockDim.x + threadIdx.x;
  long long stride = (long long)gridDim.x * blockDim.x;
  for (; i < n; i += stride) out[i] = tobf(in[i]);
}

// ------------------- embedding gather + LayerNorm (bf16 out) ---------------
__global__ __launch_bounds__(256) void embed_ln_kernel(const int* __restrict__ x,
                                                       const float* __restrict__ wemb,
                                                       const float* __restrict__ pemb,
                                                       const float* __restrict__ temb,
                                                       const float* __restrict__ lnw,
                                                       const float* __restrict__ lnb,
                                                       __bf16* __restrict__ out) {
  int row = blockIdx.x;                 // 0 .. B*S-1
  int s   = row & (S_ - 1);
  int tid = threadIdx.x;                // 256 threads, 2 elements each (HID=512)
  int tok = x[row];

  float v0 = wemb[(size_t)tok * HID_ + tid]       + pemb[(size_t)s * HID_ + tid]       + temb[tid];
  float v1 = wemb[(size_t)tok * HID_ + tid + 256] + pemb[(size_t)s * HID_ + tid + 256] + temb[tid + 256];

  float lsum = v0 + v1;
  float lsq  = v0 * v0 + v1 * v1;
  for (int off = 16; off > 0; off >>= 1) {      // wave32 reduce
    lsum += __shfl_down(lsum, off, 32);
    lsq  += __shfl_down(lsq,  off, 32);
  }
  __shared__ float ss[8], sq[8], stats[2];
  int lane = tid & 31, wid = tid >> 5;
  if (lane == 0) { ss[wid] = lsum; sq[wid] = lsq; }
  __syncthreads();
  if (tid == 0) {
    float s1 = 0.f, s2 = 0.f;
    for (int i = 0; i < 8; ++i) { s1 += ss[i]; s2 += sq[i]; }
    float mean = s1 * (1.0f / HID_);
    float var  = s2 * (1.0f / HID_) - mean * mean;
    stats[0] = mean;
    stats[1] = rsqrtf(var + 1e-12f);
  }
  __syncthreads();
  float mean = stats[0], inv = stats[1];
  out[(size_t)row * HID_ + tid]       = tobf((v0 - mean) * inv * lnw[tid]       + lnb[tid]);
  out[(size_t)row * HID_ + tid + 256] = tobf((v1 - mean) * inv * lnw[tid + 256] + lnb[tid + 256]);
}

// ---------------------------------------------------------------------------
// Tiled WMMA GEMM:  C[M,N] = A(MxK) * B(KxN), B operand given as Bt rows (NxK)
// unless TB (then stored [K,N]); A stored [M,K] unless TA (then [K,M]).
// BLOCK 128x128x32, 256 threads = 8 wave32 waves in a 4(M) x 2(N) grid,
// each wave computes 2x4 WMMA 16x16 fragments. LDS is double buffered; the
// non-transposed paths stage via GLOBAL_LOAD_ASYNC_TO_LDS_B128 so the next
// K-tile is in flight while the current one feeds the WMMAs.
// EPI: 0 = store bf16 ; 1 = bf16((acc+bias)^2) ; 2 = f32 store acc+bias
// ---------------------------------------------------------------------------
template <bool TA, bool TB, int EPI>
__global__ __launch_bounds__(256) void wmma_gemm_kernel(
    const __bf16* __restrict__ A, const __bf16* __restrict__ B,
    void* __restrict__ C, const float* __restrict__ bias,
    int M, int N, int K, int lda, int ldb, int ldc,
    int zdiv,
    long long aOuter, long long aInner,
    long long bOuter, long long bInner,
    long long cOuter, long long cInner,
    int biasMod, int biasInner) {
  (void)M; (void)N;
  __shared__ __bf16 As[2][128 * 40];
  __shared__ __bf16 Bs[2][128 * 40];

  const int tid  = threadIdx.x;
  const int lane = tid & 31;
  const int wid  = tid >> 5;
  const int wm   = (wid & 3) * 32;     // wave M offset in tile
  const int wn   = (wid >> 2) * 64;    // wave N offset in tile
  const int rowBase = blockIdx.y * 128;
  const int colBase = blockIdx.x * 128;

  // batch decomposition (z = b*NH + h for the seq-mix launches)
  const int z  = blockIdx.z;
  const long long zo = z / zdiv, zi = z % zdiv;
  A += zo * aOuter + zi * aInner;
  B += zo * bOuter + zi * bInner;
  char* Cb = (char*)C;
  Cb += (zo * cOuter + zi * cInner) * (EPI == 2 ? 4 : 2);
  const int biasBase = (z % biasMod) * biasInner;

  // ---- stage one K-tile (32 wide) into LDS buffer `buf` ----
  auto stageA = [&](int buf, int k0) {
    if constexpr (!TA) {
#pragma unroll
      for (int c = 0; c < 2; ++c) {
        int lin = tid * 16 + c * 8;
        int m = lin >> 5, kk = lin & 31;
        async_copy_b128(A + (size_t)(rowBase + m) * lda + (size_t)(k0 + kk),
                        &As[buf][m * 40 + kk]);
      }
    } else {
#pragma unroll
      for (int c = 0; c < 2; ++c) {
        int lin = tid * 16 + c * 8;
        int kk = lin >> 7, m = lin & 127;
        const __bf16* src = A + (size_t)(k0 + kk) * lda + (size_t)(rowBase + m);
        v8bf v = *(const v8bf*)src;
        __builtin_prefetch(src + (size_t)32 * lda, 0, 1);   // speculative
#pragma unroll
        for (int e = 0; e < 8; ++e) As[buf][(m + e) * 40 + kk] = v[e];
      }
    }
  };
  auto stageB = [&](int buf, int k0) {
    if constexpr (!TB) {
#pragma unroll
      for (int c = 0; c < 2; ++c) {
        int lin = tid * 16 + c * 8;
        int n = lin >> 5, kk = lin & 31;
        async_copy_b128(B + (size_t)(colBase + n) * ldb + (size_t)(k0 + kk),
                        &Bs[buf][n * 40 + kk]);
      }
    } else {
#pragma unroll
      for (int c = 0; c < 2; ++c) {
        int lin = tid * 16 + c * 8;
        int kk = lin >> 7, n = lin & 127;
        const __bf16* src = B + (size_t)(k0 + kk) * ldb + (size_t)(colBase + n);
        v8bf v = *(const v8bf*)src;
        __builtin_prefetch(src + (size_t)32 * ldb, 0, 1);   // speculative
#pragma unroll
        for (int e = 0; e < 8; ++e) Bs[buf][(n + e) * 40 + kk] = v[e];
      }
    }
  };

  const v8f vzero = {};
  v8f acc[2][4];
#pragma unroll
  for (int i = 0; i < 2; ++i)
#pragma unroll
    for (int j = 0; j < 4; ++j) acc[i][j] = vzero;

  // prologue: tile 0 into buffer 0
  stageA(0, 0);
  stageB(0, 0);

  for (int k0 = 0; k0 < K; k0 += 32) {
    const int cur = (k0 >> 5) & 1;
    if constexpr (!TA || !TB) wait_async0();   // my async writes for `cur` done
    __syncthreads();                           // everyone's tile `cur` visible
    if (k0 + 32 < K) {                         // kick off next tile into other buf
      stageA(cur ^ 1, k0 + 32);
      stageB(cur ^ 1, k0 + 32);
    }

    // ---- fragment loads (ISA 16-bit A/B VGPR layouts) ----
    const int arow = lane & 15, alo = (lane >> 4) * 8;
    v16bf af[2];
#pragma unroll
    for (int i = 0; i < 2; ++i) {
      const __bf16* p = &As[cur][(wm + 16 * i + arow) * 40 + alo];
      v8bf lo = *(const v8bf*)p;
      v8bf hi = *(const v8bf*)(p + 16);
      af[i] = __builtin_shufflevector(lo, hi, 0, 1, 2, 3, 4, 5, 6, 7,
                                              8, 9, 10, 11, 12, 13, 14, 15);
    }
    const int bcol = lane & 15, bko = (lane >> 4) * 8;
    v16bf bfr[4];
#pragma unroll
    for (int j = 0; j < 4; ++j) {
      const __bf16* p = &Bs[cur][(wn + 16 * j + bcol) * 40 + bko];
      v8bf lo = *(const v8bf*)p;
      v8bf hi = *(const v8bf*)(p + 16);
      bfr[j] = __builtin_shufflevector(lo, hi, 0, 1, 2, 3, 4, 5, 6, 7,
                                               8, 9, 10, 11, 12, 13, 14, 15);
    }
#pragma unroll
    for (int i = 0; i < 2; ++i)
#pragma unroll
      for (int j = 0; j < 4; ++j)
        acc[i][j] = __builtin_amdgcn_wmma_f32_16x16x32_bf16(
            false, af[i], false, bfr[j], (short)0, acc[i][j], false, false);
  }

  // ---- epilogue (C/D layout: VGPR r -> M = r + 8*(lane>=16), N = lane&15) ----
  const int erow = (lane >> 4) * 8;
  const int ecol = lane & 15;
#pragma unroll
  for (int i = 0; i < 2; ++i)
#pragma unroll
    for (int j = 0; j < 4; ++j)
#pragma unroll
      for (int r = 0; r < 8; ++r) {
        int gr = rowBase + wm + 16 * i + erow + r;
        int gc = colBase + wn + 16 * j + ecol;
        float v = acc[i][j][r];
        if constexpr (EPI == 0) {
          ((__bf16*)Cb)[(size_t)gr * ldc + gc] = tobf(v);
        } else if constexpr (EPI == 1) {
          float t = v + bias[biasBase + gc];
          ((__bf16*)Cb)[(size_t)gr * ldc + gc] = tobf(t * t);
        } else {
          ((float*)Cb)[(size_t)gr * ldc + gc] = v + bias[biasBase + gc];
        }
      }
}

// ---------------------------------------------------------------------------
extern "C" void kernel_launch(void* const* d_in, const int* in_sizes, int n_in,
                              void* d_out, int out_size, void* d_ws, size_t ws_size,
                              hipStream_t stream) {
  (void)in_sizes; (void)n_in; (void)out_size; (void)ws_size;

  const int*   x        = (const int*)d_in[0];
  const float* word_emb = (const float*)d_in[1];
  const float* pos_emb  = (const float*)d_in[2];
  const float* tok_emb  = (const float*)d_in[3];
  const float* ln_w     = (const float*)d_in[4];
  const float* ln_b     = (const float*)d_in[5];
  const float* init_d   = (const float*)d_in[6];
  const float* init_b   = (const float*)d_in[7];
  const float* init_M   = (const float*)d_in[8];
  const float* inter0_d = (const float*)d_in[9];
  const float* inter0_b = (const float*)d_in[10];
  const float* inter0_M = (const float*)d_in[11];
  const float* final_d  = (const float*)d_in[12];
  const float* final_b  = (const float*)d_in[13];
  float* out = (float*)d_out;

  // ---- workspace carve-up (all sizes multiples of 256B) ----
  char* p = (char*)d_ws;
  __bf16* T0  = (__bf16*)p; p += (size_t)B_ * S_ * HID_ * 2;        //  8 MB act0
  __bf16* T1  = (__bf16*)p; p += (size_t)B_ * S_ * INTER_ * 2;      // 32 MB ping
  __bf16* T2  = (__bf16*)p; p += (size_t)B_ * S_ * INTER_ * 2;      // 32 MB pong
  __bf16* Wd1 = (__bf16*)p; p += (size_t)INTER_ * HID_ * 2;         // init_d bf16
  __bf16* Wd2 = (__bf16*)p; p += (size_t)INTER_ * INTER_ * 2;       // inter0_d bf16
  __bf16* Wd3 = (__bf16*)p; p += (size_t)HID_ * INTER_ * 2;         // final_d bf16
  __bf16* M1  = (__bf16*)p; p += (size_t)NH_ * S_ * S_ * 2;         // init_M bf16
  __bf16* M2  = (__bf16*)p; p += (size_t)NH_ * S_ * S_ * 2;         // inter0_M bf16

  // ---- one-time fp32 -> bf16 weight conversion ----
  f32_to_bf16_kernel<<<2048, 256, 0, stream>>>(init_d,   Wd1, (long long)INTER_ * HID_);
  f32_to_bf16_kernel<<<4096, 256, 0, stream>>>(inter0_d, Wd2, (long long)INTER_ * INTER_);
  f32_to_bf16_kernel<<<2048, 256, 0, stream>>>(final_d,  Wd3, (long long)HID_ * INTER_);
  f32_to_bf16_kernel<<<4096, 256, 0, stream>>>(init_M,   M1,  (long long)NH_ * S_ * S_);
  f32_to_bf16_kernel<<<4096, 256, 0, stream>>>(inter0_M, M2,  (long long)NH_ * S_ * S_);

  // ---- embeddings + LayerNorm -> bf16 activations ----
  embed_ln_kernel<<<B_ * S_, 256, 0, stream>>>(x, word_emb, pos_emb, tok_emb, ln_w, ln_b, T0);

  const long long SS  = (long long)S_ * S_;
  const long long SI  = (long long)S_ * INTER_;
  dim3 blk(256);

  // GEMM1: (8192x512) * init_d^T -> T1 (8192x2048) bf16
  wmma_gemm_kernel<false, false, 0><<<dim3(INTER_ / 128, (B_ * S_) / 128, 1), blk, 0, stream>>>(
      T0, Wd1, T1, nullptr, B_ * S_, INTER_, HID_, HID_, HID_, INTER_,
      1, 0, 0, 0, 0, 0, 0, 1, 0);

  // mix1: per (b,h): C[t,d] = sum_s M1[h][s][t] * T1[b][s][h*256+d]; +bias; square
  wmma_gemm_kernel<true, true, 1><<<dim3(2, S_ / 128, B_ * NH_), blk, 0, stream>>>(
      M1, T1, T2, init_b, S_, INTER_ / NH_, S_, S_, INTER_, INTER_,
      NH_, 0, SS, SI, INTER_ / NH_, SI, INTER_ / NH_, NH_, INTER_ / NH_);

  // GEMM2: T2 * inter0_d^T -> T1 (8192x2048) bf16
  wmma_gemm_kernel<false, false, 0><<<dim3(INTER_ / 128, (B_ * S_) / 128, 1), blk, 0, stream>>>(
      T2, Wd2, T1, nullptr, B_ * S_, INTER_, INTER_, INTER_, INTER_, INTER_,
      1, 0, 0, 0, 0, 0, 0, 1, 0);

  // mix2: same structure with inter0_M / inter0_b
  wmma_gemm_kernel<true, true, 1><<<dim3(2, S_ / 128, B_ * NH_), blk, 0, stream>>>(
      M2, T1, T2, inter0_b, S_, INTER_ / NH_, S_, S_, INTER_, INTER_,
      NH_, 0, SS, SI, INTER_ / NH_, SI, INTER_ / NH_, NH_, INTER_ / NH_);

  // GEMM3: T2 * final_d^T + final_b -> d_out (8192x512) f32
  wmma_gemm_kernel<false, false, 2><<<dim3(HID_ / 128, (B_ * S_) / 128, 1), blk, 0, stream>>>(
      T2, Wd3, out, final_b, B_ * S_, HID_, INTER_, INTER_, INTER_, HID_,
      1, 0, 0, 0, 0, 0, 0, 1, 0);
}
